// MLPCaster_54022098649280
// MI455X (gfx1250) — matrix operand order repaired and checked by reading, not compiled, not run on hardware
//
#include <hip/hip_runtime.h>
#include <hip/hip_bf16.h>

// ---------------------------------------------------------------------------
// MLPCaster fused pipeline for MI455X (gfx1250, wave32, WMMA).
//   GEMM1 [N,936]x[936,64] dominates (15.7 GFLOP) -> v_wmma_f32_16x16x32_f16.
//   K padded per-joint 39->40 => K=960 = 3 chunks x 320K (8 joints/chunk).
//   Features computed once per (row,k) into LDS (packed f16 pairs), A frag =
//   2x ds_load_b128; B weights stored K-contiguous per column, B frag =
//   2x global_load_b128.
//   ws layout: hbuf (N*24 f32) | w0pT (64*480 u32) | w1pT (32*32 u32)
// ---------------------------------------------------------------------------

typedef __attribute__((ext_vector_type(16))) _Float16 v16h;
typedef __attribute__((ext_vector_type(8)))  float    v8f;
typedef __attribute__((ext_vector_type(4)))  unsigned u32x4;
typedef __attribute__((ext_vector_type(2)))  __fp16   fp16x2;  // cvt_pkrtz ret type

#define JN    24
#define HID   64
#define KPAD  960
#define KPT   (KPAD / 2)      /* 480 k-pairs total      */
#define KPCH  160             /* k-pairs per chunk      */
#define FSTR  164             /* LDS row stride (dwords, padded) */

union AFrag { v16h v; _Float16 h[16]; unsigned u[8]; u32x4 q[2]; };
union HPack { fp16x2 h; unsigned u; };

__device__ __forceinline__ unsigned pk(float a, float b) {
  HPack p; p.h = __builtin_amdgcn_cvt_pkrtz(a, b); return p.u;
}

// --------------------------- weight repack ---------------------------------
// w0pT[col*480 + kp] = pack(f16 w0[k0,col], f16 w0[k1,col]) with per-joint
// 39->40 zero padding; w1pT[col*32 + kp] likewise (cols padded 24->32).
__global__ __launch_bounds__(256) void pack_weights_kernel(
    const float* __restrict__ w0, const float* __restrict__ w1,
    unsigned* __restrict__ w0pT, unsigned* __restrict__ w1pT) {
  int idx = blockIdx.x * 256 + threadIdx.x;
  if (idx < KPT * 64) {
    int kp = idx >> 6, n = idx & 63;
    int j = kp / 20, fp = kp - 20 * j;       // 20 pairs per padded joint
    int f0 = 2 * fp;
    float v0 = w0[(j * 39 + f0) * 64 + n];   // f0 <= 38 always
    float v1 = (f0 + 1 >= 39) ? 0.f : w0[(j * 39 + f0 + 1) * 64 + n];
    w0pT[n * KPT + kp] = pk(v0, v1);
  } else if (idx < KPT * 64 + 32 * 32) {
    int i2 = idx - KPT * 64;
    int kp = i2 >> 5, n = i2 & 31;
    float v0 = (n < JN) ? w1[(2 * kp) * JN + n] : 0.f;
    float v1 = (n < JN) ? w1[(2 * kp + 1) * JN + n] : 0.f;
    w1pT[n * 32 + kp] = pk(v0, v1);
  }
}

// --------------------- fused encode + 2-layer MLP --------------------------
// block = 256 thr = 8 waves = 64 rows; wave (rowtile = w>>1, colhalf = w&1)
// owns 16 rows x 32 cols of GEMM1. grid = N/64.
__global__ __launch_bounds__(256) void mlp_kernel(
    const float* __restrict__ xyz, const float* __restrict__ transforms,
    const unsigned* __restrict__ w0pT, const unsigned* __restrict__ w1pT,
    float* __restrict__ hbuf, int N) {
  __shared__ float tl[JN * 16];
  __shared__ __align__(16) unsigned featp[64 * FSTR];  // 41984 B
  float* hlds = (float*)featp;                         // aliased after GEMM1

  int tid = threadIdx.x;
  for (int i = tid; i < JN * 16; i += 256) tl[i] = transforms[i];

  int w = tid >> 5, lane = tid & 31;
  int laneLo = lane & 15, laneHi = lane >> 4;
  int rowt = w >> 1, colh = w & 1;
  int rowL = rowt * 16 + laneLo;

  // compute-phase assignment: row = tid&63, k-group g = tid>>6 (2 joints each)
  int crow = tid & 63, g = tid >> 6;
  int growC = blockIdx.x * 64 + crow;
  float cx = xyz[growC * 3 + 0], cy = xyz[growC * 3 + 1], cz = xyz[growC * 3 + 2];

  v8f acc[2] = {};
  __syncthreads();  // tl ready

  for (int c = 0; c < 3; ++c) {
    // ---- feature compute: 2 whole joints per thread, all indices static ----
#pragma unroll
    for (int jj = 0; jj < 2; ++jj) {
      int j = c * 8 + g * 2 + jj;
      const float* T = &tl[j * 16];
      float a0 = fmaf(T[0], cx, fmaf(T[1], cy, fmaf(T[2], cz, T[3])));
      float a1 = fmaf(T[4], cx, fmaf(T[5], cy, fmaf(T[6], cz, T[7])));
      float a2 = fmaf(T[8], cx, fmaf(T[9], cy, fmaf(T[10], cz, T[11])));
      float lx = fmaf(a0 + 1.5f, 0.66666666667f, -1.0f);  // normalize_coord
      float ly = fmaf(a1 + 1.5f, 0.66666666667f, -1.0f);
      float lz = fmaf(a2 + 1.5f, 0.66666666667f, -1.0f);

      float vals[40];
      vals[0] = lx; vals[1] = ly; vals[2] = lz;
      float sx = lx, sy = ly, sz = lz;
#pragma unroll
      for (int fi = 0; fi < 6; ++fi) {
        int b = 3 + 6 * fi;
        vals[b + 0] = __sinf(sx); vals[b + 1] = __sinf(sy); vals[b + 2] = __sinf(sz);
        vals[b + 3] = __cosf(sx); vals[b + 4] = __cosf(sy); vals[b + 5] = __cosf(sz);
        sx += sx; sy += sy; sz += sz;      // 2^fi scaling
      }
      vals[39] = 0.0f;                     // joint pad

      unsigned* dst = &featp[crow * FSTR + g * 40 + jj * 20];
#pragma unroll
      for (int m = 0; m < 5; ++m) {        // 5x ds_store_b128
        u32x4 q = { pk(vals[8 * m + 0], vals[8 * m + 1]),
                    pk(vals[8 * m + 2], vals[8 * m + 3]),
                    pk(vals[8 * m + 4], vals[8 * m + 5]),
                    pk(vals[8 * m + 6], vals[8 * m + 7]) };
        *(u32x4*)(dst + 4 * m) = q;
      }
    }
    __syncthreads();

    // ---- GEMM1: 10 K-steps of 32 over this chunk ----
    for (int kk = 0; kk < 10; ++kk) {
      // A 16x32 f16: interleaved kp pattern == two contiguous 4-dword runs
      const unsigned* ar = &featp[rowL * FSTR + kk * 16 + laneHi * 4];
      AFrag a;
      a.q[0] = *(const u32x4*)ar;          // kp base+{0..3|4..7}
      a.q[1] = *(const u32x4*)(ar + 8);    // kp base+{8..11|12..15}

      int kpg = c * KPCH + kk * 16 + laneHi * 4;
      if (kk + 1 < 10)                     // global_prefetch_b8 next B line
        __builtin_prefetch(&w0pT[(colh * 32 + laneLo) * KPT + kpg + 16], 0, 3);
#pragma unroll
      for (int ct = 0; ct < 2; ++ct) {
        int col = colh * 32 + ct * 16 + laneLo;
        const unsigned* br = &w0pT[col * KPT + kpg];
        AFrag b;                           // B 32x16: lane-half holds 16 K
        b.q[0] = *(const u32x4*)br;
        b.q[1] = *(const u32x4*)(br + 8);
        acc[ct] = __builtin_amdgcn_wmma_f32_16x16x32_f16(
            false, a.v, false, b.v, (short)0, acc[ct], false, false);
      }
    }
    __syncthreads();  // featp reusable for next chunk
  }

  // ---- relu(h1) -> LDS transpose bounce (featp memory reused) ----
#pragma unroll
  for (int ct = 0; ct < 2; ++ct)
#pragma unroll
    for (int gg = 0; gg < 8; ++gg)
      hlds[(rowt * 16 + gg + laneHi * 8) * HID + colh * 32 + ct * 16 + laneLo] =
          fmaxf(acc[ct][gg], 0.0f);
  __syncthreads();

  // ---- GEMM2: [16,64]x[64,32pad], waves 0..3 (one rowtile each) ----
  if (w < 4) {
    v8f c2[2] = {};
#pragma unroll
    for (int kt = 0; kt < 2; ++kt) {
      AFrag a2;
#pragma unroll
      for (int i = 0; i < 8; ++i) {
        int kp = kt * 16 + ((i & 3) | ((i >> 2) << 3) | (laneHi << 2));
        float2 f2 = *(const float2*)&hlds[(w * 16 + laneLo) * HID + 2 * kp];
        a2.u[i] = pk(f2.x, f2.y);
      }
#pragma unroll
      for (int ct = 0; ct < 2; ++ct) {
        const unsigned* br = &w1pT[(ct * 16 + laneLo) * 32 + kt * 16 + laneHi * 4];
        AFrag b2;
        b2.q[0] = *(const u32x4*)br;
        b2.q[1] = *(const u32x4*)(br + 8);
        c2[ct] = __builtin_amdgcn_wmma_f32_16x16x32_f16(
            false, a2.v, false, b2.v, (short)0, c2[ct], false, false);
      }
    }
    // sigma = relu(h2) stored row-major [N,24]; the faithful view(J,-1).T
    // scramble is then just hbuf[j*N + n] in the warp kernel.
    int rowbase = blockIdx.x * 64 + w * 16;
#pragma unroll
    for (int ct = 0; ct < 2; ++ct)
#pragma unroll
      for (int gg = 0; gg < 8; ++gg) {
        int col = ct * 16 + laneLo;
        if (col < JN)
          hbuf[(rowbase + gg + laneHi * 8) * JN + col] = fmaxf(c2[ct][gg], 0.0f);
      }
  }
}

// --------------------------- warp / blend ----------------------------------
__global__ __launch_bounds__(256) void warp_kernel(
    const float* __restrict__ xyz, const float* __restrict__ vd,
    const float* __restrict__ transforms, const float* __restrict__ hbuf,
    float* __restrict__ out, int N) {
  __shared__ float tl[JN * 16];
  int tid = threadIdx.x;
  for (int i = tid; i < JN * 16; i += 256) tl[i] = transforms[i];
  __syncthreads();

  int n = blockIdx.x * 256 + tid;
  float w[JN];
  float s = 0.f;
#pragma unroll
  for (int j = 0; j < JN; ++j) {  // weights[n,j] = relu_h_flat[j*N+n]
    w[j] = hbuf[j * N + n];
    s += w[j];
  }
  float inv = 1.0f / (s + 1e-6f);

  float A[12] = {};
#pragma unroll
  for (int j = 0; j < JN; ++j) {
    float wn = w[j] * inv;
    const float* T = &tl[j * 16];
#pragma unroll
    for (int e = 0; e < 12; ++e) A[e] = fmaf(wn, T[e], A[e]);
  }

  float px = xyz[n * 3 + 0], py = xyz[n * 3 + 1], pz = xyz[n * 3 + 2];
  float vx = vd[n * 3 + 0],  vy = vd[n * 3 + 1],  vz = vd[n * 3 + 2];

  float ox = fmaf(A[0], px, fmaf(A[1], py, fmaf(A[2], pz, A[3])));
  float oy = fmaf(A[4], px, fmaf(A[5], py, fmaf(A[6], pz, A[7])));
  float oz = fmaf(A[8], px, fmaf(A[9], py, fmaf(A[10], pz, A[11])));
  // vd_warped = out[:N]-out[N:] == Sum_j wn_j * (T_j[:3,:3] @ vd)
  float dx = fmaf(A[0], vx, fmaf(A[1], vy, A[2] * vz));
  float dy = fmaf(A[4], vx, fmaf(A[5], vy, A[6] * vz));
  float dz = fmaf(A[8], vx, fmaf(A[9], vy, A[10] * vz));

  out[n * 3 + 0] = ox;
  out[n * 3 + 1] = oy;
  out[n * 3 + 2] = oz;
  out[N * 3 + n * 3 + 0] = dx;
  out[N * 3 + n * 3 + 1] = dy;
  out[N * 3 + n * 3 + 2] = dz;
}

// ---------------------------------------------------------------------------
extern "C" void kernel_launch(void* const* d_in, const int* in_sizes, int n_in,
                              void* d_out, int out_size, void* d_ws, size_t ws_size,
                              hipStream_t stream) {
  const float* xyz        = (const float*)d_in[0];  // [R,S,3]
  const float* viewdirs   = (const float*)d_in[1];  // [R,S,3]
  const float* transforms = (const float*)d_in[2];  // [24,4,4]
  // d_in[3] = ray_valid (all true, unused by the reference math)
  const float* w0         = (const float*)d_in[4];  // [936,64]
  const float* w1         = (const float*)d_in[5];  // [64,24]

  int N = in_sizes[0] / 3;  // 131072

  char* ws = (char*)d_ws;
  float*    hbuf = (float*)ws;                             // N*24 f32
  unsigned* w0pT = (unsigned*)(ws + (size_t)N * JN * 4);   // 64*480 u32
  unsigned* w1pT = w0pT + 64 * KPT;                        // 32*32  u32

  pack_weights_kernel<<<(KPT * 64 + 32 * 32 + 255) / 256, 256, 0, stream>>>(
      w0, w1, w0pT, w1pT);
  mlp_kernel<<<N / 64, 256, 0, stream>>>(xyz, transforms, w0pT, w1pT, hbuf, N);
  warp_kernel<<<N / 256, 256, 0, stream>>>(xyz, viewdirs, transforms, hbuf,
                                           (float*)d_out, N);
}
